// RPNHead_12335146074309
// MI455X (gfx1250) — compile-verified, matrix operand order checked by reading, not run on hardware
//
#include <hip/hip_runtime.h>
#include <hip/hip_bf16.h>
#include <math.h>

// ---------------------------------------------------------------------------
// RPN head for MI455X (gfx1250, wave32, WMMA).
// conv3x3 as implicit GEMM: M = B*H*W = 30400, N = 512, K = C*9 = 4608,
// bf16 A/B with f32 accumulate via v_wmma_f32_16x16x32_bf16.
// Staging uses async global->LDS (ASYNCcnt) with double buffering when the
// toolchain declares the gfx1250 async builtins; otherwise falls back to
// global_load_b128 + ds_store_b128.
// ---------------------------------------------------------------------------

typedef __attribute__((ext_vector_type(16))) __bf16 v16bf;
typedef __attribute__((ext_vector_type(8)))  float  v8f;
typedef __attribute__((ext_vector_type(4)))  int    v4i;

#define BB   8
#define CC   512
#define HH   50
#define WWW  76
#define HWW  3800           // H*W
#define MM   30400          // B*H*W
#define KK   4608           // C*9
#define NH   64             // head N (54 used, padded to 64)

#ifndef __has_builtin
#define __has_builtin(x) 0
#endif

#if __has_builtin(__builtin_amdgcn_global_load_async_to_lds_b128)
#define HAVE_ASYNC 1
#else
#define HAVE_ASYNC 0
#endif

#if HAVE_ASYNC
typedef __attribute__((address_space(1))) v4i* g_b128_p;   // global int4*
typedef __attribute__((address_space(3))) v4i* l_b128_p;   // LDS int4*
#endif

__device__ __forceinline__ void async_b128(const __bf16* g, __bf16* l) {
#if HAVE_ASYNC
    __builtin_amdgcn_global_load_async_to_lds_b128(
        (g_b128_p)(g), (l_b128_p)(l), 0, 0);
#else
    *(uint4*)l = *(const uint4*)g;
#endif
}

template <int N>
__device__ __forceinline__ void wait_async() {
#if HAVE_ASYNC
#if __has_builtin(__builtin_amdgcn_s_wait_asynccnt)
    __builtin_amdgcn_s_wait_asynccnt(N);
#else
    asm volatile("s_wait_asynccnt %0" :: "i"(N));
#endif
#endif
}

__device__ __forceinline__ __bf16 f2bf(float f) {
    unsigned u = __builtin_bit_cast(unsigned, f);
    u += 0x7fffu + ((u >> 16) & 1u);          // round-to-nearest-even
    unsigned short s = (unsigned short)(u >> 16);
    return __builtin_bit_cast(__bf16, s);
}

// ---------------- prep kernels (bandwidth trivial) -------------------------

// NCHW f32 -> NHWC bf16
__global__ void prep_feat(const float* __restrict__ src, __bf16* __restrict__ dst) {
    int idx = blockIdx.x * blockDim.x + threadIdx.x;
    if (idx >= BB * CC * HH * WWW) return;
    int x = idx % WWW;
    int y = (idx / WWW) % HH;
    int c = (idx / HWW) % CC;
    int b = idx / (HWW * CC);
    dst[((size_t)((b * HH + y) * WWW + x)) * CC + c] = f2bf(src[idx]);
}

// conv_w[o][c][ky][kx] f32 -> Wt[o][k] bf16, k = (ky*3+kx)*512 + c
__global__ void prep_convw(const float* __restrict__ src, __bf16* __restrict__ dst) {
    int idx = blockIdx.x * blockDim.x + threadIdx.x;
    if (idx >= CC * CC * 9) return;
    int j = idx % 9;
    int c = (idx / 9) % CC;
    int o = idx / (9 * CC);
    dst[(size_t)o * KK + j * CC + c] = f2bf(src[idx]);
}

// pack cls_w(18x512) and reg_w(36x512) into Wh[64][512] bf16 (rows 54..63 zero)
__global__ void prep_headw(const float* __restrict__ cls_w,
                           const float* __restrict__ reg_w,
                           __bf16* __restrict__ dst) {
    int idx = blockIdx.x * blockDim.x + threadIdx.x;
    if (idx >= NH * CC) return;
    int k = idx % CC;
    int n = idx / CC;
    float v = 0.f;
    if (n < 18)       v = cls_w[n * CC + k];
    else if (n < 54)  v = reg_w[(n - 18) * CC + k];
    dst[idx] = f2bf(v);
}

// ---------------- conv 3x3 implicit GEMM, WMMA bf16 ------------------------

union Frag { v16bf v; uint4 q[2]; };

#define A_ELEMS (64 * 32)     // one A buffer: 64 rows x 32 bf16
#define B_ELEMS (256 * 32)    // one B buffer: 256 rows x 32 bf16

__global__ __launch_bounds__(256)
void conv3x3_wmma(const __bf16* __restrict__ feat,   // NHWC bf16
                  const __bf16* __restrict__ wt,     // [O][K]
                  const float*  __restrict__ bias,   // [512]
                  __bf16*       __restrict__ out)    // [M][512] bf16 (post ReLU)
{
    __shared__ __align__(16) __bf16 ldsA[2 * A_ELEMS];   //  8 KB
    __shared__ __align__(16) __bf16 ldsB[2 * B_ELEMS];   // 32 KB

    const int tid   = threadIdx.x;
    const int mBase = blockIdx.x * 64;     // 475 tiles, exact
    const int nBase = blockIdx.y * 256;    // 2 tiles

    const int wv   = tid >> 5;
    const int lane = tid & 31;
    const int wm   = wv >> 2;              // 0..1  (M half, 32 rows)
    const int wn   = wv & 3;               // 0..3  (N quarter, 64 cols)
    const int lr   = lane & 15;
    const int lh   = (lane >> 4) << 3;     // 0 or 8 (K-chunk select)

    // A-staging coordinates (per thread, constant over K loop)
    const int ar   = tid >> 2;             // 0..63 row
    const int aoff = (tid & 3) << 3;       // 0,8,16,24 elements
    const int am   = mBase + ar;
    const int ab   = am / HWW;
    const int arem = am - ab * HWW;
    const int ay   = arem / WWW;
    const int ax   = arem - ay * WWW;

    v8f acc[2][4];
    #pragma unroll
    for (int mi = 0; mi < 2; ++mi)
        #pragma unroll
        for (int ni = 0; ni < 4; ++ni)
            #pragma unroll
            for (int r = 0; r < 8; ++r)
                acc[mi][ni][r] = 0.f;

    // stage one K-step (32 contiguous K) into buffer `buf`
    auto stage = [&](int ks, int buf) {
        const int j  = ks >> 4;            // filter tap 0..8
        const int kc = (ks & 15) << 5;     // c offset 0..480
        const int dy = j / 3 - 1;
        const int dx = j % 3 - 1;
        // A: clamp OOB rows to a safe in-bounds address (zero-patched later)
        int ys = ay + dy, xs = ax + dx;
        int ysC = min(max(ys, 0), HH - 1);
        int xsC = min(max(xs, 0), WWW - 1);
        const __bf16* pa = feat +
            ((size_t)((ab * HH + ysC) * WWW + xsC)) * CC + kc + aoff;
        async_b128(pa, &ldsA[buf * A_ELEMS + ar * 32 + aoff]);
        // B: 256 rows x 64B, 4 b128 per thread
        #pragma unroll
        for (int t = 0; t < 4; ++t) {
            const int idx = tid + (t << 8);
            const int n   = idx >> 2;
            const int off = (idx & 3) << 3;
            async_b128(wt + (size_t)(nBase + n) * KK + j * CC + kc + off,
                       &ldsB[buf * B_ELEMS + n * 32 + off]);
        }
    };

    stage(0, 0);

    for (int ks = 0; ks < 144; ++ks) {
        const int cur = ks & 1;
        if (ks + 1 < 144) {
            stage(ks + 1, cur ^ 1);
            wait_async<5>();               // drain batch `cur` (in-order), keep next 5 in flight
        } else {
            wait_async<0>();
        }
        // zero-patch SAME-padding rows of the A tile we are about to consume
        {
            const int j  = ks >> 4;
            const int ys = ay + j / 3 - 1;
            const int xs = ax + j % 3 - 1;
            if (!((unsigned)ys < (unsigned)HH && (unsigned)xs < (unsigned)WWW))
                *(uint4*)&ldsA[cur * A_ELEMS + ar * 32 + aoff] =
                    make_uint4(0u, 0u, 0u, 0u);
        }
        __syncthreads();

        // fragments follow the documented 16-bit A/B VGPR striping
        Frag fa[2], fb[4];
        #pragma unroll
        for (int mi = 0; mi < 2; ++mi) {
            const int row = wm * 32 + mi * 16 + lr;
            fa[mi].q[0] = *(const uint4*)&ldsA[cur * A_ELEMS + row * 32 + lh];
            fa[mi].q[1] = *(const uint4*)&ldsA[cur * A_ELEMS + row * 32 + lh + 16];
        }
        #pragma unroll
        for (int ni = 0; ni < 4; ++ni) {
            const int row = wn * 64 + ni * 16 + lr;
            fb[ni].q[0] = *(const uint4*)&ldsB[cur * B_ELEMS + row * 32 + lh];
            fb[ni].q[1] = *(const uint4*)&ldsB[cur * B_ELEMS + row * 32 + lh + 16];
        }
        #pragma unroll
        for (int mi = 0; mi < 2; ++mi)
            #pragma unroll
            for (int ni = 0; ni < 4; ++ni)
                acc[mi][ni] = __builtin_amdgcn_wmma_f32_16x16x32_bf16(
                    false, fa[mi].v, false, fb[ni].v,
                    (short)0, acc[mi][ni], false, false);

        __syncthreads();
    }

    // epilogue: bias + ReLU, write bf16 NHWC for head GEMM
    #pragma unroll
    for (int mi = 0; mi < 2; ++mi)
        #pragma unroll
        for (int ni = 0; ni < 4; ++ni)
            #pragma unroll
            for (int r = 0; r < 8; ++r) {
                const int m = mBase + wm * 32 + mi * 16 + r + ((lane >> 4) << 3);
                const int n = nBase + wn * 64 + ni * 16 + (lane & 15);
                float v = acc[mi][ni][r] + bias[n];
                v = v > 0.f ? v : 0.f;
                out[(size_t)m * CC + n] = f2bf(v);
            }
}

// ---------------- head GEMM: [M x 512] * [512 x 64] ------------------------

#define HA_ELEMS (64 * 32)
#define HB_ELEMS (64 * 32)

__global__ __launch_bounds__(256)
void head_gemm_wmma(const __bf16* __restrict__ fin,   // [M][512] bf16
                    const __bf16* __restrict__ wh,    // [64][512] bf16
                    const float*  __restrict__ cls_b, // [18]
                    const float*  __restrict__ reg_b, // [36]
                    float*        __restrict__ ho)    // [M][64] f32
{
    __shared__ __align__(16) __bf16 ldsA[2 * HA_ELEMS];
    __shared__ __align__(16) __bf16 ldsB[2 * HB_ELEMS];

    const int tid   = threadIdx.x;
    const int mBase = blockIdx.x * 64;   // 475 tiles, exact

    const int wv   = tid >> 5;
    const int lane = tid & 31;
    const int mi   = wv & 3;             // M subtile per wave
    const int nh   = (wv >> 2) * 32;     // N half per wave
    const int lr   = lane & 15;
    const int lh   = (lane >> 4) << 3;

    const int sr   = tid >> 2;
    const int soff = (tid & 3) << 3;

    v8f acc[2];
    #pragma unroll
    for (int ni = 0; ni < 2; ++ni)
        #pragma unroll
        for (int r = 0; r < 8; ++r)
            acc[ni][r] = 0.f;

    auto stage = [&](int ks, int buf) {
        const int kc = ks << 5;
        async_b128(&fin[(size_t)(mBase + sr) * CC + kc + soff],
                   &ldsA[buf * HA_ELEMS + sr * 32 + soff]);
        async_b128(&wh[(size_t)sr * CC + kc + soff],
                   &ldsB[buf * HB_ELEMS + sr * 32 + soff]);
    };

    stage(0, 0);

    for (int ks = 0; ks < 16; ++ks) {
        const int cur = ks & 1;
        if (ks + 1 < 16) { stage(ks + 1, cur ^ 1); wait_async<2>(); }
        else             { wait_async<0>(); }
        __syncthreads();

        Frag fa, fb[2];
        {
            const int row = mi * 16 + lr;
            fa.q[0] = *(const uint4*)&ldsA[cur * HA_ELEMS + row * 32 + lh];
            fa.q[1] = *(const uint4*)&ldsA[cur * HA_ELEMS + row * 32 + lh + 16];
        }
        #pragma unroll
        for (int ni = 0; ni < 2; ++ni) {
            const int row = nh + ni * 16 + lr;
            fb[ni].q[0] = *(const uint4*)&ldsB[cur * HB_ELEMS + row * 32 + lh];
            fb[ni].q[1] = *(const uint4*)&ldsB[cur * HB_ELEMS + row * 32 + lh + 16];
        }
        #pragma unroll
        for (int ni = 0; ni < 2; ++ni)
            acc[ni] = __builtin_amdgcn_wmma_f32_16x16x32_bf16(
                false, fa.v, false, fb[ni].v, (short)0, acc[ni], false, false);

        __syncthreads();
    }

    #pragma unroll
    for (int ni = 0; ni < 2; ++ni)
        #pragma unroll
        for (int r = 0; r < 8; ++r) {
            const int m = mBase + mi * 16 + r + ((lane >> 4) << 3);
            const int n = nh + ni * 16 + (lane & 15);
            float b = 0.f;
            if (n < 18)      b = cls_b[n];
            else if (n < 54) b = reg_b[n - 18];
            ho[(size_t)m * NH + n] = acc[ni][r] + b;
        }
}

// ---------------- anchor decode (elementwise) ------------------------------

__global__ void decode_kernel(const float* __restrict__ ho,       // [M][64]
                              const int*   __restrict__ img_size, // [B][2]
                              float*       __restrict__ d_out)
{
    const int idx = blockIdx.x * blockDim.x + threadIdx.x;   // anchor index
    if (idx >= MM * 9) return;
    const int a = idx % 9;
    const int m = idx / 9;                 // = b*3800 + y*76 + x
    const int x = m % WWW;
    const int y = (m / WWW) % HH;
    const int b = m / HWW;

    const float ratios[3] = {0.5f, 1.0f, 2.0f};
    const float scales[3] = {128.0f, 256.0f, 512.0f};
    const float rat = ratios[a / 3];
    const float scl = scales[a % 3];
    const float sq  = sqrtf(rat);
    const float aw  = scl / sq;
    const float ah  = scl * sq;
    const float axc = (float)(x * 16) + 8.0f;
    const float ayc = (float)(y * 16) + 8.0f;

    const float* t = ho + (size_t)m * NH + 18 + a * 4;
    const float px = axc + t[0] * aw;
    const float py = ayc + t[1] * ah;
    const float pw = aw * expf(t[2]);
    const float ph = ah * expf(t[3]);

    const float imw = (float)img_size[b * 2 + 0];
    const float imh = (float)img_size[b * 2 + 1];

    float x1 = px - pw * 0.5f, y1 = py - ph * 0.5f;
    float x2 = x1 + pw,        y2 = y1 + ph;
    x1 = fminf(fmaxf(x1, 0.f), imw);  y1 = fminf(fmaxf(y1, 0.f), imh);
    x2 = fminf(fmaxf(x2, 0.f), imw);  y2 = fminf(fmaxf(y2, 0.f), imh);
    const float bw = x2 - x1, bh = y2 - y1;

    float4 prop = make_float4(x1 + bw * 0.5f, y1 + bh * 0.5f, bw, bh);
    *(float4*)&d_out[(size_t)idx * 4] = prop;

    // cls logits (bias already applied in head GEMM)
    float* cls_out = d_out + (size_t)MM * 9 * 4;
    cls_out[(size_t)idx * 2 + 0] = ho[(size_t)m * NH + a * 2 + 0];
    cls_out[(size_t)idx * 2 + 1] = ho[(size_t)m * NH + a * 2 + 1];
}

// ---------------- launcher -------------------------------------------------

extern "C" void kernel_launch(void* const* d_in, const int* in_sizes, int n_in,
                              void* d_out, int out_size, void* d_ws, size_t ws_size,
                              hipStream_t stream) {
    const float* feat   = (const float*)d_in[0];
    const float* conv_w = (const float*)d_in[1];
    const float* conv_b = (const float*)d_in[2];
    const float* cls_w  = (const float*)d_in[3];
    const float* cls_b  = (const float*)d_in[4];
    const float* reg_w  = (const float*)d_in[5];
    const float* reg_b  = (const float*)d_in[6];
    const int*   imgsz  = (const int*)  d_in[7];

    // workspace layout (all offsets 256B aligned)
    char* ws = (char*)d_ws;
    const size_t SZ_FEAT = (size_t)MM * CC * 2;        // 31,129,600
    const size_t SZ_WT   = (size_t)CC * KK * 2;        //  4,718,592
    const size_t SZ_FOUT = (size_t)MM * CC * 2;        // 31,129,600
    const size_t SZ_WH   = (size_t)NH * CC * 2;        //     65,536
    __bf16* featN = (__bf16*)(ws);
    __bf16* wt    = (__bf16*)(ws + SZ_FEAT);
    __bf16* fout  = (__bf16*)(ws + SZ_FEAT + SZ_WT);
    __bf16* wh    = (__bf16*)(ws + SZ_FEAT + SZ_WT + SZ_FOUT);
    float*  ho    = (float*) (ws + SZ_FEAT + SZ_WT + SZ_FOUT + SZ_WH);

    // 1) precision/layout transforms
    prep_feat <<<(BB * CC * HH * WWW + 255) / 256, 256, 0, stream>>>(feat, featN);
    prep_convw<<<(CC * CC * 9 + 255) / 256,        256, 0, stream>>>(conv_w, wt);
    prep_headw<<<(NH * CC + 255) / 256,            256, 0, stream>>>(cls_w, reg_w, wh);

    // 2) conv3x3 implicit GEMM (compute bound, WMMA bf16, async double-buffer)
    conv3x3_wmma<<<dim3(MM / 64, CC / 256), 256, 0, stream>>>(featN, wt, conv_b, fout);

    // 3) fused cls/reg head GEMM
    head_gemm_wmma<<<MM / 64, 256, 0, stream>>>(fout, wh, cls_b, reg_b, ho);

    // 4) anchor decode
    decode_kernel<<<(MM * 9 + 255) / 256, 256, 0, stream>>>(ho, imgsz, (float*)d_out);
}
